// GraphEncoderAttentionWithNeighborAttentionResidual_6219112645398
// MI455X (gfx1250) — matrix operand order repaired and checked by reading, not compiled, not run on hardware
//
#include <hip/hip_runtime.h>
#include <hip/hip_bf16.h>
#include <stdint.h>

typedef __attribute__((ext_vector_type(16))) _Float16 v16h;
typedef __attribute__((ext_vector_type(8)))  float    v8f;

#define Nn 20000
#define Ee 320000
#define Hh 256
#define EN (Ee + Nn)
#define NHIDn 600
#define NOUTn 768
#define Gg 256
#define MT (Nn / 16)   /* 1250 row tiles */
#define NTt (Hh / 16)  /* 16 col tiles  */
#define NWAVES ((MT / 2) * 4) /* 2500 macro-tile waves */

// ---------------------------------------------------------------- utilities
__global__ void k_fill(float* p, float v, int n) {
  int i = blockIdx.x * blockDim.x + threadIdx.x;
  if (i < n) p[i] = v;
}

__device__ __forceinline__ void atomicMaxF(float* addr, float val) {
  if (val >= 0.0f) atomicMax((int*)addr, __float_as_int(val));
  else             atomicMin((unsigned int*)addr, __float_as_uint(val));
}

// ---------------------------------------------------- degree / GCN norm
__global__ void k_degacc(const long long* __restrict__ ei, float* deg) {
  int e = blockIdx.x * blockDim.x + threadIdx.x;
  if (e < Ee) atomicAdd(&deg[(int)ei[Ee + e]], 1.0f);
}
__global__ void k_rsqrt_inplace(float* p, int n) {
  int i = blockIdx.x * blockDim.x + threadIdx.x;
  if (i < n) p[i] = rsqrtf(p[i]);
}
__global__ void k_norm(const long long* __restrict__ ei, const float* __restrict__ dinv,
                       float* __restrict__ nrm) {
  int e = blockIdx.x * blockDim.x + threadIdx.x;
  if (e >= EN) return;
  int s, d;
  if (e < Ee) { s = (int)ei[e]; d = (int)ei[Ee + e]; } else { s = d = e - Ee; }
  nrm[e] = dinv[s] * dinv[d];
}

// ------------------------------------------------ WMMA operand packing
// A fragment layout (16-bit A, 16x32, wave32; ISA 7.12.2):
//  lane = L, m = L&15, sel = L>>4; element e: k = kk*32 + sel*8 + e (e<8)
//                                             k = kk*32 + 16 + sel*8 + (e-8) (e>=8)
__global__ void k_packA(const float* __restrict__ X, _Float16* __restrict__ AP) {
  int gid = blockIdx.x * blockDim.x + threadIdx.x;           // MT*8*32 threads
  if (gid >= MT * 8 * 32) return;
  int lane = gid & 31, kk = (gid >> 5) & 7, mt = gid >> 8;
  int m = mt * 16 + (lane & 15), sel = lane >> 4;
  const float* row = X + (size_t)m * Hh + kk * 32 + sel * 8;
  _Float16* dp = AP + (size_t)gid * 16;
#pragma unroll
  for (int e = 0; e < 8; ++e) dp[e] = (_Float16)row[e];
#pragma unroll
  for (int e = 0; e < 8; ++e) dp[8 + e] = (_Float16)row[16 + e];
}
// B fragment (16-bit B, 32x16): lane n = L&15, sel = L>>4; element e: k = sel*8 + e (e<8), 16 + sel*8 + (e-8) (e>=8)
__global__ void k_packB(const float* __restrict__ W, _Float16* __restrict__ BP) {
  int gid = blockIdx.x * blockDim.x + threadIdx.x;           // 16*8*32 threads
  if (gid >= NTt * 8 * 32) return;
  int lane = gid & 31, kk = (gid >> 5) & 7, nt = gid >> 8;
  int n = nt * 16 + (lane & 15), sel = lane >> 4;
  _Float16* dp = BP + (size_t)gid * 16;
#pragma unroll
  for (int e = 0; e < 16; ++e) {
    int k = kk * 32 + (e < 8 ? sel * 8 + e : 16 + sel * 8 + (e - 8));
    dp[e] = (_Float16)W[(size_t)k * Hh + n];
  }
}

// -------------------------------------------- WMMA GEMM: C[N,H] = A[N,256] * W[256,H]
// Register-blocked: each wave computes a 32x64 macro-tile (2 M-tiles x 4 N-tiles,
// 8 f32 accumulators). Per K-step: 2 A-frag + 4 B-frag loads feed 8 v_wmma.
__global__ void k_gemm(const _Float16* __restrict__ AP, const _Float16* __restrict__ BP,
                       float* __restrict__ C) {
  int w = blockIdx.x * (blockDim.x >> 5) + (threadIdx.x >> 5);
  if (w >= NWAVES) return;
  int lane = threadIdx.x & 31;
  int mt0 = (w >> 2) * 2;
  int nt0 = (w & 3) * 4;
  v8f c00 = {}, c01 = {}, c02 = {}, c03 = {};
  v8f c10 = {}, c11 = {}, c12 = {}, c13 = {};
#pragma unroll
  for (int kk = 0; kk < 8; ++kk) {
    v16h a0 = *(const v16h*)(AP + ((size_t)((mt0 + 0) * 8 + kk) * 32 + lane) * 16);
    v16h a1 = *(const v16h*)(AP + ((size_t)((mt0 + 1) * 8 + kk) * 32 + lane) * 16);
    v16h b0 = *(const v16h*)(BP + ((size_t)((nt0 + 0) * 8 + kk) * 32 + lane) * 16);
    v16h b1 = *(const v16h*)(BP + ((size_t)((nt0 + 1) * 8 + kk) * 32 + lane) * 16);
    v16h b2 = *(const v16h*)(BP + ((size_t)((nt0 + 2) * 8 + kk) * 32 + lane) * 16);
    v16h b3 = *(const v16h*)(BP + ((size_t)((nt0 + 3) * 8 + kk) * 32 + lane) * 16);
    c00 = __builtin_amdgcn_wmma_f32_16x16x32_f16(false, a0, false, b0, (short)0, c00, false, false);
    c10 = __builtin_amdgcn_wmma_f32_16x16x32_f16(false, a1, false, b0, (short)0, c10, false, false);
    c01 = __builtin_amdgcn_wmma_f32_16x16x32_f16(false, a0, false, b1, (short)0, c01, false, false);
    c11 = __builtin_amdgcn_wmma_f32_16x16x32_f16(false, a1, false, b1, (short)0, c11, false, false);
    c02 = __builtin_amdgcn_wmma_f32_16x16x32_f16(false, a0, false, b2, (short)0, c02, false, false);
    c12 = __builtin_amdgcn_wmma_f32_16x16x32_f16(false, a1, false, b2, (short)0, c12, false, false);
    c03 = __builtin_amdgcn_wmma_f32_16x16x32_f16(false, a0, false, b3, (short)0, c03, false, false);
    c13 = __builtin_amdgcn_wmma_f32_16x16x32_f16(false, a1, false, b3, (short)0, c13, false, false);
  }
  int n = lane & 15, sel = lane >> 4;
  auto st = [&](const v8f& c, int mt, int nt) {
    float* cp = C + (size_t)(mt * 16 + sel * 8) * Hh + nt * 16 + n;
#pragma unroll
    for (int r = 0; r < 8; ++r) cp[(size_t)r * Hh] = c[r];
  };
  st(c00, mt0 + 0, nt0 + 0); st(c01, mt0 + 0, nt0 + 1);
  st(c02, mt0 + 0, nt0 + 2); st(c03, mt0 + 0, nt0 + 3);
  st(c10, mt0 + 1, nt0 + 0); st(c11, mt0 + 1, nt0 + 1);
  st(c12, mt0 + 1, nt0 + 2); st(c13, mt0 + 1, nt0 + 3);
}

// ------------------------------------------------ message passing
__global__ void k_init_bias(float* __restrict__ out, const float* __restrict__ bias) {
  int i = blockIdx.x * blockDim.x + threadIdx.x;
  if (i < Nn * Hh) out[i] = bias[i & (Hh - 1)];
}
// out[dst] += H[src] * coef[e]  over E+N edges; 64 threads/edge, float4 per thread
__global__ void k_scatter(const float* __restrict__ Hm, const long long* __restrict__ ei,
                          const float* __restrict__ coef, float* __restrict__ out) {
  int e = blockIdx.x * 4 + (threadIdx.x >> 6);
  if (e >= EN) return;
  int s, d;
  if (e < Ee) { s = (int)ei[e]; d = (int)ei[Ee + e]; } else { s = d = e - Ee; }
  float cf = coef[e];
  int f = (threadIdx.x & 63) * 4;
  float4 v = *(const float4*)(Hm + (size_t)s * Hh + f);
  float* o = out + (size_t)d * Hh + f;
  atomicAdd(o + 0, v.x * cf);
  atomicAdd(o + 1, v.y * cf);
  atomicAdd(o + 2, v.z * cf);
  atomicAdd(o + 3, v.w * cf);
}

// ------------------------------------------------ batch norm
__global__ void k_bnstats(const float* __restrict__ X, float* s, float* s2) {
  int c = threadIdx.x;
  float a = 0.f, b = 0.f;
  for (int r = blockIdx.x; r < Nn; r += gridDim.x) {
    float v = X[(size_t)r * Hh + c];
    a += v; b += v * v;
  }
  atomicAdd(&s[c], a);
  atomicAdd(&s2[c], b);
}
__global__ void k_bnfin(const float* s, const float* s2, const float* __restrict__ g,
                        const float* __restrict__ be, float* scl, float* shf) {
  int c = threadIdx.x;
  float mu = s[c] * (1.0f / Nn);
  float var = s2[c] * (1.0f / Nn) - mu * mu;
  float sc = g[c] * rsqrtf(var + 1e-5f);
  scl[c] = sc;
  shf[c] = be[c] - mu * sc;
}
__global__ void k_bnrelures(const float* __restrict__ A, const float* __restrict__ scl,
                            const float* __restrict__ shf, const float* __restrict__ xres,
                            float* __restrict__ xout) {
  int i = blockIdx.x * blockDim.x + threadIdx.x;
  if (i >= Nn * Hh) return;
  int c = i & (Hh - 1);
  float v = A[i] * scl[c] + shf[c];
  v = v > 0.f ? v : 0.f;
  xout[i] = v + xres[i];
}

// ------------------------------------------------ GAT attention
__global__ void k_gatdots(const float* __restrict__ Hm, const float* __restrict__ av,
                          const float* __restrict__ aw, float* ss, float* sd) {
  int t = blockIdx.x * blockDim.x + threadIdx.x;
  int node = t >> 5, lane = t & 31;
  if (node >= Nn) return;
  const float* hp = Hm + (size_t)node * Hh + lane * 8;
  float s1 = 0.f, s2 = 0.f;
#pragma unroll
  for (int j = 0; j < 8; ++j) {
    float v = hp[j];
    s1 += v * av[lane * 8 + j];
    s2 += v * aw[lane * 8 + j];
  }
#pragma unroll
  for (int off = 16; off; off >>= 1) {
    s1 += __shfl_xor(s1, off, 32);
    s2 += __shfl_xor(s2, off, 32);
  }
  if (lane == 0) { ss[node] = s1; sd[node] = s2; }
}
__global__ void k_scoremax(const long long* __restrict__ ei, const float* __restrict__ ss,
                           const float* __restrict__ sd, float* eb, float* mmax) {
  int e = blockIdx.x * blockDim.x + threadIdx.x;
  if (e >= EN) return;
  int s, d;
  if (e < Ee) { s = (int)ei[e]; d = (int)ei[Ee + e]; } else { s = d = e - Ee; }
  float v = ss[s] + sd[d];
  v = v > 0.f ? v : 0.2f * v;   // leaky_relu(.,0.2)
  eb[e] = v;
  atomicMaxF(&mmax[d], v);
}
__global__ void k_expdenom(const long long* __restrict__ ei, const float* __restrict__ mmax,
                           float* eb, float* denom) {
  int e = blockIdx.x * blockDim.x + threadIdx.x;
  if (e >= EN) return;
  int d = (e < Ee) ? (int)ei[Ee + e] : e - Ee;
  float ex = __expf(eb[e] - mmax[d]);
  eb[e] = ex;
  atomicAdd(&denom[d], ex);
}
__global__ void k_alphanorm(const long long* __restrict__ ei, const float* __restrict__ denom,
                            float* eb) {
  int e = blockIdx.x * blockDim.x + threadIdx.x;
  if (e >= EN) return;
  int d = (e < Ee) ? (int)ei[Ee + e] : e - Ee;
  eb[e] = eb[e] / denom[d];
}

// ------------------------------------------------ pooling + MLP
__global__ void k_poolsum(const float* __restrict__ X, const long long* __restrict__ batch,
                          float* sums) {
  int t = blockIdx.x * blockDim.x + threadIdx.x;
  int node = t >> 6;
  if (node >= Nn) return;
  int f = (t & 63) * 4;
  float4 v = *(const float4*)(X + (size_t)node * Hh + f);
  int g = (int)batch[node];
  float* o = sums + (size_t)g * Hh + f;
  atomicAdd(o + 0, v.x);
  atomicAdd(o + 1, v.y);
  atomicAdd(o + 2, v.z);
  atomicAdd(o + 3, v.w);
}
__global__ void k_poolcnt(const long long* __restrict__ batch, float* cnt) {
  int i = blockIdx.x * blockDim.x + threadIdx.x;
  if (i < Nn) atomicAdd(&cnt[(int)batch[i]], 1.0f);
}
__global__ void k_poolfin(float* sums, const float* cnt) {
  int i = blockIdx.x * blockDim.x + threadIdx.x;
  if (i >= Gg * Hh) return;
  sums[i] /= fmaxf(cnt[i >> 8], 1.0f);
}
__global__ void k_mlp1(const float* __restrict__ pooled, const float* __restrict__ W,
                       const float* __restrict__ b, float* __restrict__ hid) {
  __shared__ float sh[Hh];
  int g = blockIdx.x;
  sh[threadIdx.x] = pooled[(size_t)g * Hh + threadIdx.x];
  __syncthreads();
  int j = blockIdx.y * 256 + threadIdx.x;
  if (j >= NHIDn) return;
  float acc = b[j];
  for (int k = 0; k < Hh; ++k) acc += sh[k] * W[(size_t)k * NHIDn + j];
  hid[(size_t)g * NHIDn + j] = fmaxf(acc, 0.f);
}
__global__ void k_mlp2(const float* __restrict__ hid, const float* __restrict__ W,
                       const float* __restrict__ b, float* __restrict__ out) {
  __shared__ float sh[NHIDn];
  int g = blockIdx.x;
  for (int k = threadIdx.x; k < NHIDn; k += 256) sh[k] = hid[(size_t)g * NHIDn + k];
  __syncthreads();
  int j = blockIdx.y * 256 + threadIdx.x;
  if (j >= NOUTn) return;
  float acc = b[j];
  for (int k = 0; k < NHIDn; ++k) acc += sh[k] * W[(size_t)k * NOUTn + j];
  out[(size_t)g * NOUTn + j] = acc;
}

// ================================================================ launcher
extern "C" void kernel_launch(void* const* d_in, const int* in_sizes, int n_in,
                              void* d_out, int out_size, void* d_ws, size_t ws_size,
                              hipStream_t stream) {
  (void)in_sizes; (void)n_in; (void)out_size; (void)ws_size;
  const float*     x     = (const float*)d_in[0];
  const long long* ei    = (const long long*)d_in[1];
  const long long* batch = (const long long*)d_in[2];
  const float *W1 = (const float*)d_in[3],  *b1 = (const float*)d_in[4],
              *g1 = (const float*)d_in[5],  *be1 = (const float*)d_in[6],
              *W2 = (const float*)d_in[7],  *b2 = (const float*)d_in[8],
              *g2 = (const float*)d_in[9],  *be2 = (const float*)d_in[10],
              *W3 = (const float*)d_in[11], *b3 = (const float*)d_in[12],
              *g3 = (const float*)d_in[13], *be3 = (const float*)d_in[14],
              *Wa = (const float*)d_in[15], *ba = (const float*)d_in[16],
              *asv = (const float*)d_in[17], *adv = (const float*)d_in[18],
              *ga = (const float*)d_in[19], *bea = (const float*)d_in[20],
              *Wm1 = (const float*)d_in[21], *bm1 = (const float*)d_in[22],
              *Wm2 = (const float*)d_in[23], *bm2 = (const float*)d_in[24];

  char* ws = (char*)d_ws;
  size_t cur = 0;
  auto alloc = [&](size_t bytes) -> void* {
    void* p = ws + cur;
    cur = (cur + bytes + 255) & ~(size_t)255;
    return p;
  };
  float*    B0    = (float*)alloc(sizeof(float) * Nn * Hh);
  float*    B1    = (float*)alloc(sizeof(float) * Nn * Hh);
  float*    AG    = (float*)alloc(sizeof(float) * Nn * Hh);
  _Float16* AP    = (_Float16*)alloc(sizeof(_Float16) * Nn * Hh);
  _Float16* BP    = (_Float16*)alloc(sizeof(_Float16) * Hh * Hh);
  float*    nrm   = (float*)alloc(sizeof(float) * EN);
  float*    eb    = (float*)alloc(sizeof(float) * EN);
  float*    deg   = (float*)alloc(sizeof(float) * Nn);   // becomes dinv in-place
  float*    ss    = (float*)alloc(sizeof(float) * Nn);
  float*    sd    = (float*)alloc(sizeof(float) * Nn);
  float*    mmax  = (float*)alloc(sizeof(float) * Nn);
  float*    denom = (float*)alloc(sizeof(float) * Nn);
  float*    csum  = (float*)alloc(sizeof(float) * 2 * Hh);
  float*    csq   = csum + Hh;
  float*    scl   = (float*)alloc(sizeof(float) * Hh);
  float*    shf   = (float*)alloc(sizeof(float) * Hh);
  float*    sums  = (float*)alloc(sizeof(float) * Gg * Hh);
  float*    cnt   = (float*)alloc(sizeof(float) * Gg);
  float*    hid   = (float*)alloc(sizeof(float) * Gg * NHIDn);

  auto g1d = [](int n, int t) { return (n + t - 1) / t; };

  // degree / symmetric norm (self-loops contribute the initial 1.0)
  k_fill<<<g1d(Nn, 256), 256, 0, stream>>>(deg, 1.0f, Nn);
  k_degacc<<<g1d(Ee, 256), 256, 0, stream>>>(ei, deg);
  k_rsqrt_inplace<<<g1d(Nn, 256), 256, 0, stream>>>(deg, Nn);
  k_norm<<<g1d(EN, 256), 256, 0, stream>>>(ei, deg, nrm);

  auto gemm = [&](const float* xin, const float* W, float* hbuf) {
    k_packB<<<NTt, 256, 0, stream>>>(W, BP);
    k_packA<<<MT, 256, 0, stream>>>(xin, AP);
    k_gemm<<<g1d(NWAVES, 8), 256, 0, stream>>>(AP, BP, hbuf);
  };
  auto bn_relu_res = [&](const float* gv, const float* bev, const float* xres, float* xout) {
    k_fill<<<2, 256, 0, stream>>>(csum, 0.0f, 2 * Hh);
    k_bnstats<<<256, 256, 0, stream>>>(AG, csum, csq);
    k_bnfin<<<1, 256, 0, stream>>>(csum, csq, gv, bev, scl, shf);
    k_bnrelures<<<g1d(Nn * Hh, 256), 256, 0, stream>>>(AG, scl, shf, xres, xout);
  };
  auto gcn = [&](const float* xin, const float* W, const float* bv, const float* gv,
                 const float* bev, float* hbuf, float* xout) {
    gemm(xin, W, hbuf);
    k_init_bias<<<g1d(Nn * Hh, 256), 256, 0, stream>>>(AG, bv);
    k_scatter<<<g1d(EN, 4), 256, 0, stream>>>(hbuf, ei, nrm, AG);
    bn_relu_res(gv, bev, xin, xout);
  };

  // three GCN layers
  gcn(x,  W1, b1, g1, be1, B0, B1);   // B1 = layer1 out
  gcn(B1, W2, b2, g2, be2, B0, B0);   // hbuf=B0, xout=B0 (hbuf dead by then)
  gcn(B0, W3, b3, g3, be3, B1, B1);   // B1 = layer3 out

  // GAT layer: h = B1 @ Wa  -> B0
  gemm(B1, Wa, B0);
  k_gatdots<<<g1d(Nn * 32, 256), 256, 0, stream>>>(B0, asv, adv, ss, sd);
  k_fill<<<g1d(Nn, 256), 256, 0, stream>>>(mmax, -__builtin_inff(), Nn);
  k_scoremax<<<g1d(EN, 256), 256, 0, stream>>>(ei, ss, sd, eb, mmax);
  k_fill<<<g1d(Nn, 256), 256, 0, stream>>>(denom, 0.0f, Nn);
  k_expdenom<<<g1d(EN, 256), 256, 0, stream>>>(ei, mmax, eb, denom);
  k_alphanorm<<<g1d(EN, 256), 256, 0, stream>>>(ei, denom, eb);
  k_init_bias<<<g1d(Nn * Hh, 256), 256, 0, stream>>>(AG, ba);
  k_scatter<<<g1d(EN, 4), 256, 0, stream>>>(B0, ei, eb, AG);
  bn_relu_res(ga, bea, B1, B0);       // final node features in B0

  // global mean pool
  k_fill<<<g1d(Gg * Hh, 256), 256, 0, stream>>>(sums, 0.0f, Gg * Hh);
  k_fill<<<1, 256, 0, stream>>>(cnt, 0.0f, Gg);
  k_poolsum<<<g1d(Nn * 64, 256), 256, 0, stream>>>(B0, batch, sums);
  k_poolcnt<<<g1d(Nn, 256), 256, 0, stream>>>(batch, cnt);
  k_poolfin<<<g1d(Gg * Hh, 256), 256, 0, stream>>>(sums, cnt);

  // MLP head
  k_mlp1<<<dim3(Gg, 3), 256, 0, stream>>>(sums, Wm1, bm1, hid);
  k_mlp2<<<dim3(Gg, 3), 256, 0, stream>>>(hid, Wm2, bm2, (float*)d_out);
}
